// Attention_35622458753806
// MI455X (gfx1250) — matrix operand order repaired and checked by reading, not compiled, not run on hardware
//
#include <hip/hip_runtime.h>

// ---------------------------------------------------------------------------
// MI455X (gfx1250) attention block, bf16 WMMA pipeline + async LDS staging.
// B=4, S=2048, E=P=1024, H=8, D=128.
// ---------------------------------------------------------------------------

#define B_  4
#define S_  2048
#define E_  1024
#define P_  1024
#define H_  8
#define D_  128
#define SCALE_ 0.35355339059327373f   // 1/sqrt(H) per reference (not 1/sqrt(D)!)

typedef __attribute__((ext_vector_type(16))) __bf16 bf16x16;
typedef __attribute__((ext_vector_type(8)))  float  f32x8;

union FragB16 {
    bf16x16 v;
    uint4   q[2];
    unsigned short u[16];
};

__device__ __forceinline__ unsigned short f2bf(float f) {
    unsigned int u = __float_as_uint(f);
    unsigned int r = u + 0x7FFFu + ((u >> 16) & 1u);   // round-to-nearest-even
    return (unsigned short)(r >> 16);
}

__device__ __forceinline__ f32x8 wmma_bf16(bf16x16 a, bf16x16 b, f32x8 c) {
    return __builtin_amdgcn_wmma_f32_16x16x32_bf16(
        false, a, false, b, (short)0, c, false, false);
}

// 32-byte-contiguous B-style fragment (col n = lane&15, 16 contiguous k)
__device__ __forceinline__ void load_frag_contig(FragB16& f, const unsigned short* p) {
    f.q[0] = *reinterpret_cast<const uint4*>(p);
    f.q[1] = *reinterpret_cast<const uint4*>(p + 8);
}

// gfx1250 async global->LDS copy (VGLOBAL encoding, ASYNCcnt-tracked).
// Generic pointers to __shared__ carry the LDS byte offset in bits[31:0].
__device__ __forceinline__ void async_load_b128(const void* gptr, void* lptr) {
    unsigned lds_off = (unsigned)(size_t)lptr;
    asm volatile("global_load_async_to_lds_b128 %0, %1, off"
                 :: "v"(lds_off), "v"(gptr) : "memory");
}
__device__ __forceinline__ void wait_async() {
    asm volatile("s_wait_asynccnt 0" ::: "memory");
}

// ---------------------------------------------------------------------------
// fp32 -> bf16 conversion
// ---------------------------------------------------------------------------
__global__ void f32_to_bf16_k(const float* __restrict__ s,
                              unsigned short* __restrict__ d, int n) {
    int i = blockIdx.x * blockDim.x + threadIdx.x;
    if (i < n) d[i] = f2bf(s[i]);
}

// ---------------------------------------------------------------------------
// Tiled bf16 GEMM:  C[M,N] (f32) = A[M,K] (bf16, row major) x B[N,K]^T (bf16)
// Block: 256 threads = 8 waves (2 M x 4 N); wave tile 32x64; block 64x256.
// K step 32.  Requires M%64==0, N%256==0, K%32==0 (true here).
// ---------------------------------------------------------------------------
__global__ __launch_bounds__(256) void gemm_bf16_k(
    const unsigned short* __restrict__ A,
    const unsigned short* __restrict__ Bw,
    float* __restrict__ C, int M, int N, int K)
{
    __shared__ __align__(16) unsigned short Al[64][40];    // 32 k + 8 pad
    __shared__ __align__(16) unsigned short Bl[256][40];
    const int tid  = threadIdx.x;
    const int lane = tid & 31;
    const int w    = tid >> 5;
    const int wm   = w >> 2;          // 0..1
    const int wn   = w & 3;           // 0..3
    const int m0   = blockIdx.y * 64;
    const int n0   = blockIdx.x * 256;
    const int lr   = lane & 15;
    const int lcA  = (lane >> 4) * 16;  // A fragment k base (16 contiguous)
    const int lcB  = (lane >> 4) * 16;  // B fragment k base (16 contiguous)

    const int ra = tid >> 2,  ca = (tid & 3) * 8;   // A staging: 64x32

    f32x8 acc[2][4] = {};

    for (int kk = 0; kk < K; kk += 32) {
        // stage A: 64 x 32 via async DMA to LDS
        async_load_b128(A + (size_t)(m0 + ra) * K + kk + ca, &Al[ra][ca]);
        // stage B: 256 x 32
        #pragma unroll
        for (int i = 0; i < 4; ++i) {
            int cidx = tid + i * 256;
            int r = cidx >> 2, c8 = (cidx & 3) * 8;
            async_load_b128(Bw + (size_t)(n0 + r) * K + kk + c8, &Bl[r][c8]);
        }
        if (kk + 32 < K) {   // L2 prefetch of next B tile slice
            __builtin_prefetch(Bw + (size_t)(n0 + ra) * K + kk + 32 + ca, 0, 1);
        }
        wait_async();
        __syncthreads();

        FragB16 af[2], bfr[4];
        #pragma unroll
        for (int mi = 0; mi < 2; ++mi) {
            int row = wm * 32 + mi * 16 + lr;
            af[mi].q[0] = *reinterpret_cast<const uint4*>(&Al[row][lcA]);
            af[mi].q[1] = *reinterpret_cast<const uint4*>(&Al[row][8 + lcA]);
        }
        #pragma unroll
        for (int ni = 0; ni < 4; ++ni) {
            int col = wn * 64 + ni * 16 + lr;
            load_frag_contig(bfr[ni], &Bl[col][lcB]);
        }
        #pragma unroll
        for (int mi = 0; mi < 2; ++mi)
            #pragma unroll
            for (int ni = 0; ni < 4; ++ni)
                acc[mi][ni] = wmma_bf16(af[mi].v, bfr[ni].v, acc[mi][ni]);
        __syncthreads();
    }

    const int mbase = m0 + wm * 32 + ((lane < 16) ? 0 : 8);
    const int nbase = n0 + wn * 64 + lr;
    #pragma unroll
    for (int mi = 0; mi < 2; ++mi)
        #pragma unroll
        for (int ni = 0; ni < 4; ++ni)
            #pragma unroll
            for (int r = 0; r < 8; ++r)
                C[(size_t)(mbase + mi * 16 + r) * N + nbase + ni * 16] = acc[mi][ni][r];
}

// ---------------------------------------------------------------------------
// Raw-view head split [B,S,P] -> [B,H,S,D] with optional interleaved RoPE,
// fp32 in, bf16 out.  One thread per (b,h,s2,j) pair (d = 2j, 2j+1).
// Mapping: L = h*S*D + s2*D + d ; src s = L/P = h*256 + s2/8 ; p = L%P.
// ---------------------------------------------------------------------------
__global__ void rope_split_k(const float* __restrict__ src,
                             unsigned short* __restrict__ dst, int do_rope)
{
    int idx = blockIdx.x * blockDim.x + threadIdx.x;   // 2^22 total
    if (idx >= B_ * H_ * S_ * (D_ / 2)) return;
    int j  = idx & 63;
    int s2 = (idx >> 6) & (S_ - 1);
    int h  = (idx >> 17) & (H_ - 1);
    int b  = idx >> 20;
    int s  = h * 256 + (s2 >> 3);
    int p  = (s2 & 7) * D_ + 2 * j;
    const float* sp = src + ((size_t)b * S_ + s) * P_ + p;
    float x0 = sp[0], x1 = sp[1];
    float o0 = x0, o1 = x1;
    if (do_rope) {
        float inv = __powf(10000.f, -(float)(2 * j) * (1.0f / (float)D_));
        float ang = (float)s2 * inv;
        float sn, cs;
        __sincosf(ang, &sn, &cs);
        o0 = x0 * cs - x1 * sn;
        o1 = x1 * cs + x0 * sn;
    }
    unsigned short* dp = dst + (((size_t)(b * H_ + h) * S_) + s2) * D_ + 2 * j;
    dp[0] = f2bf(o0);
    dp[1] = f2bf(o1);
}

// ---------------------------------------------------------------------------
// Flash attention: per (b,h), online-softmax over S keys in blocks of 32.
// Block = 256 threads (8 waves); each wave owns 16 query rows -> 128 q/block.
// Scores and P*V both via v_wmma_f32_16x16x32_bf16.  K tile staged with
// async DMA; V staged transposed through VGPRs.
// ---------------------------------------------------------------------------
__global__ __launch_bounds__(256) void flash_attn_k(
    const unsigned short* __restrict__ Q,
    const unsigned short* __restrict__ Kd,
    const unsigned short* __restrict__ V,
    unsigned short* __restrict__ Out)
{
    __shared__ __align__(16) unsigned short Klds[32][136];   // [key][d] + pad
    __shared__ __align__(16) unsigned short Vt[128][40];     // [d][key] + pad
    __shared__ __align__(16) unsigned short Pscr[8][16][40]; // per-wave P tile

    const int tid  = threadIdx.x;
    const int lane = tid & 31;
    const int w    = tid >> 5;
    const int b    = blockIdx.z, h = blockIdx.y;
    const int qr0  = blockIdx.x * 128 + w * 16;
    const int lr   = lane & 15;
    const int lcA  = (lane >> 4) * 16;
    const int lcB  = (lane >> 4) * 16;

    const unsigned short* qb = Q  + ((size_t)(b * H_ + h) * S_) * D_;
    const unsigned short* kb = Kd + ((size_t)(b * H_ + h) * S_) * D_;
    const unsigned short* vb = V  + ((size_t)(b * H_ + h) * S_) * D_;

    FragB16 qf[4];
    #pragma unroll
    for (int c = 0; c < 4; ++c) {
        const unsigned short* p = qb + (size_t)(qr0 + lr) * D_ + 32 * c;
        qf[c].q[0] = *reinterpret_cast<const uint4*>(p + lcA);
        qf[c].q[1] = *reinterpret_cast<const uint4*>(p + 8 + lcA);
    }

    f32x8 oacc[8] = {};
    float mrow[8], lrow[8];
    #pragma unroll
    for (int r = 0; r < 8; ++r) { mrow[r] = -1e30f; lrow[r] = 0.f; }

    const int mloc = (lane < 16) ? 0 : 8;

    for (int j0 = 0; j0 < S_; j0 += 32) {
        // ---- stage K (async DMA) and V (transposed) into LDS ----
        #pragma unroll
        for (int i = 0; i < 2; ++i) {
            int c  = tid + i * 256;           // 0..511 over 32x128/8
            int kr = c >> 4;                  // key row 0..31
            int d0 = (c & 15) * 8;            // d chunk
            async_load_b128(kb + (size_t)(j0 + kr) * D_ + d0, &Klds[kr][d0]);
            union { uint4 q; unsigned short u[8]; } vu;
            vu.q = *reinterpret_cast<const uint4*>(vb + (size_t)(j0 + kr) * D_ + d0);
            #pragma unroll
            for (int e = 0; e < 8; ++e) Vt[d0 + e][kr] = vu.u[e];
        }
        wait_async();
        __syncthreads();

        // ---- scores: two 16x16 tiles (keys j0..+15 and j0+16..+31) ----
        f32x8 s0 = {}, s1 = {};
        #pragma unroll
        for (int c = 0; c < 4; ++c) {
            FragB16 b0, b1;
            load_frag_contig(b0, &Klds[lr][32 * c + lcB]);
            load_frag_contig(b1, &Klds[16 + lr][32 * c + lcB]);
            s0 = wmma_bf16(qf[c].v, b0.v, s0);
            s1 = wmma_bf16(qf[c].v, b1.v, s1);
        }

        // ---- online softmax; row r lives entirely in this lane's half ----
        #pragma unroll
        for (int r = 0; r < 8; ++r) {
            float a0 = s0[r] * SCALE_, a1 = s1[r] * SCALE_;
            float t = fmaxf(a0, a1);
            t = fmaxf(t, __shfl_xor(t, 1));
            t = fmaxf(t, __shfl_xor(t, 2));
            t = fmaxf(t, __shfl_xor(t, 4));
            t = fmaxf(t, __shfl_xor(t, 8));
            float mnew  = fmaxf(mrow[r], t);
            float alpha = __expf(mrow[r] - mnew);
            float p0 = __expf(a0 - mnew);
            float p1 = __expf(a1 - mnew);
            float rs = p0 + p1;
            rs += __shfl_xor(rs, 1);
            rs += __shfl_xor(rs, 2);
            rs += __shfl_xor(rs, 4);
            rs += __shfl_xor(rs, 8);
            lrow[r] = lrow[r] * alpha + rs;
            mrow[r] = mnew;
            #pragma unroll
            for (int c = 0; c < 8; ++c) oacc[c][r] *= alpha;
            Pscr[w][mloc + r][lr]      = f2bf(p0);
            Pscr[w][mloc + r][16 + lr] = f2bf(p1);
        }

        // ---- P (16x32) as A fragment, then P x V ----
        FragB16 pf;
        pf.q[0] = *reinterpret_cast<const uint4*>(&Pscr[w][lr][lcA]);
        pf.q[1] = *reinterpret_cast<const uint4*>(&Pscr[w][lr][8 + lcA]);
        #pragma unroll
        for (int c = 0; c < 8; ++c) {
            FragB16 vf;
            load_frag_contig(vf, &Vt[16 * c + lr][lcB]);
            oacc[c] = wmma_bf16(pf.v, vf.v, oacc[c]);
        }
        __syncthreads();
    }

    unsigned short* ob = Out + (size_t)b * S_ * P_ + (size_t)h * D_;
    #pragma unroll
    for (int r = 0; r < 8; ++r) {
        float inv = 1.0f / lrow[r];
        int row = qr0 + mloc + r;
        #pragma unroll
        for (int c = 0; c < 8; ++c)
            ob[(size_t)row * P_ + 16 * c + lr] = f2bf(oacc[c][r] * inv);
    }
}

// ---------------------------------------------------------------------------
// RowLinear input permutation: x2[b,i,j] = attn[b, (i*P+j)%S, (i*P+j)/S]
// ---------------------------------------------------------------------------
__global__ void permute_x2_k(const unsigned short* __restrict__ a,
                             unsigned short* __restrict__ x2)
{
    int idx = blockIdx.x * blockDim.x + threadIdx.x;   // 2^23
    if (idx >= B_ * S_ * P_) return;
    int j = idx & (P_ - 1);
    int i = (idx >> 10) & (S_ - 1);
    int b = idx >> 21;
    int L  = i * P_ + j;
    int sp = L & (S_ - 1);
    int pp = L >> 11;
    x2[idx] = a[((size_t)b * S_ + sp) * P_ + pp];
}

// ---------------------------------------------------------------------------
// RowLinear output inverse-shuffle: out[b,s,e] = yflat[b, e*S + s]
// ---------------------------------------------------------------------------
__global__ void final_permute_k(const float* __restrict__ y,
                                float* __restrict__ out)
{
    int idx = blockIdx.x * blockDim.x + threadIdx.x;   // 2^23
    if (idx >= B_ * S_ * E_) return;
    int e = idx & (E_ - 1);
    int s = (idx >> 10) & (S_ - 1);
    int b = idx >> 21;
    out[idx] = y[(size_t)b * S_ * E_ + (size_t)e * S_ + s];
}

// ---------------------------------------------------------------------------
extern "C" void kernel_launch(void* const* d_in, const int* in_sizes, int n_in,
                              void* d_out, int out_size, void* d_ws, size_t ws_size,
                              hipStream_t stream) {
    (void)in_sizes; (void)n_in; (void)out_size; (void)ws_size;
    const float* emb = (const float*)d_in[0];
    const float* W1  = (const float*)d_in[1];
    const float* W2  = (const float*)d_in[2];
    const float* W3  = (const float*)d_in[3];
    const float* Wo  = (const float*)d_in[4];
    float* out = (float*)d_out;

    const size_t EMB_N = (size_t)B_ * S_ * E_;    // 8388608
    const size_t W_N   = (size_t)P_ * E_;         // 1048576

    char* ws = (char*)d_ws;
    unsigned short* embB = (unsigned short*)ws;                  // 16 MiB
    unsigned short* w1B  = embB + EMB_N;
    unsigned short* w2B  = w1B + W_N;
    unsigned short* w3B  = w2B + W_N;
    unsigned short* woB  = w3B + W_N;
    float* scrF          = (float*)(woB + W_N);                  // 32 MiB, reused
    unsigned short* qB   = (unsigned short*)(scrF + EMB_N);
    unsigned short* kB   = qB + EMB_N;
    unsigned short* vB   = kB + EMB_N;
    unsigned short* attnB = vB + EMB_N;
    unsigned short* x2B  = qB;   // q no longer needed after attention

    dim3 blk(256);
    f32_to_bf16_k<<<(EMB_N + 255) / 256, blk, 0, stream>>>(emb, embB, (int)EMB_N);
    f32_to_bf16_k<<<(W_N + 255) / 256, blk, 0, stream>>>(W1, w1B, (int)W_N);
    f32_to_bf16_k<<<(W_N + 255) / 256, blk, 0, stream>>>(W2, w2B, (int)W_N);
    f32_to_bf16_k<<<(W_N + 255) / 256, blk, 0, stream>>>(W3, w3B, (int)W_N);
    f32_to_bf16_k<<<(W_N + 255) / 256, blk, 0, stream>>>(Wo, woB, (int)W_N);

    dim3 ggrid(P_ / 256, (B_ * S_) / 64);          // (4, 128)
    const int M = B_ * S_;

    gemm_bf16_k<<<ggrid, blk, 0, stream>>>(embB, w1B, scrF, M, P_, E_);
    rope_split_k<<<(B_ * H_ * S_ * (D_ / 2)) / 256, blk, 0, stream>>>(scrF, qB, 1);
    gemm_bf16_k<<<ggrid, blk, 0, stream>>>(embB, w2B, scrF, M, P_, E_);
    rope_split_k<<<(B_ * H_ * S_ * (D_ / 2)) / 256, blk, 0, stream>>>(scrF, kB, 1);
    gemm_bf16_k<<<ggrid, blk, 0, stream>>>(embB, w3B, scrF, M, P_, E_);
    rope_split_k<<<(B_ * H_ * S_ * (D_ / 2)) / 256, blk, 0, stream>>>(scrF, vB, 0);

    dim3 agrid(S_ / 128, H_, B_);                  // (16, 8, 4)
    flash_attn_k<<<agrid, blk, 0, stream>>>(qB, kB, vB, attnB);

    permute_x2_k<<<(B_ * S_ * P_) / 256, blk, 0, stream>>>(attnB, x2B);
    gemm_bf16_k<<<ggrid, blk, 0, stream>>>(x2B, woB, scrF, M, E_, P_);
    final_permute_k<<<(B_ * S_ * E_) / 256, blk, 0, stream>>>(scrF, out);
}
